// ContrastiveTokenLoss_13297218748901
// MI455X (gfx1250) — compile-verified, matrix operand order checked by reading, not run on hardware
//
#include <hip/hip_runtime.h>
#include <hip/hip_bf16.h>

// Contrastive token loss, MI455X (gfx1250, wave32).
// Memory-bound streaming reduction: 524 MB read once -> ~22us floor @ 23.3 TB/s.
// Strategy: NT b128 loads (no L2 pollution; zero reuse), speculative prefetch
// lookahead (unguarded: TH=0 prefetch is dropped on translation fail, never
// faults the wave), v_exp_f32, WMMA f32 16x16x4 wave32 cross-lane reduction,
// deterministic 2-pass sum (no float atomics).

typedef __attribute__((ext_vector_type(4)))  float    v4f;
typedef __attribute__((ext_vector_type(4)))  int      v4i;
typedef __attribute__((ext_vector_type(2)))  float    v2f;
typedef __attribute__((ext_vector_type(8)))  float    v8f;
typedef __attribute__((ext_vector_type(16))) _Float16 v16h;

#define CTL_B     4
#define CTL_S     512
#define CTL_V     32000
#define CTL_V4    (CTL_V / 4)
#define CTL_ROWS  (CTL_B * CTL_S)
#define CTL_BS    256          // 8 wave32s per block
#define CTL_PAD   (-1)

// Reduce one f32 partial per lane across the whole wave32 using a single WMMA.
// A holds each lane's partial once (other K slots zero), B is all ones, so the
// 16 row-sums of D collectively contain every lane's value exactly once.
// Summing this lane's 8 accumulator regs + xor-16 shuffle => wave total in
// every lane. Exact f32 accumulation, fixed order => deterministic.
__device__ __forceinline__ float wave_reduce_wmma(float p) {
#if __has_builtin(__builtin_amdgcn_wmma_f32_16x16x4_f32)
    v2f a; a[0] = p;    a[1] = 0.0f;
    v2f b; b[0] = 1.0f; b[1] = 1.0f;
    v8f c = {};
    c = __builtin_amdgcn_wmma_f32_16x16x4_f32(
            /*neg_a=*/false, a, /*neg_b=*/false, b,
            /*c_mod=*/(short)0, c, /*reuse_a=*/false, /*reuse_b=*/false);
#else
    // Fallback: probe-confirmed f16 WMMA (partials ~1e2, f16-safe).
    v16h a = {};
    a[0] = (_Float16)p;
    v16h b;
    #pragma unroll
    for (int i = 0; i < 16; ++i) b[i] = (_Float16)1.0f;
    v8f c = {};
    c = __builtin_amdgcn_wmma_f32_16x16x32_f16(
            false, a, false, b, (short)0, c, false, false);
#endif
    float s = c[0] + c[1] + c[2] + c[3] + c[4] + c[5] + c[6] + c[7];
    s += __shfl_xor(s, 16, 32);   // lanes 0-15 hold rows 0-7, 16-31 hold 8-15
    return s;
}

// Kernel 1: one block per (b,s) row. Streams 32000 logits + 32000 masks,
// accumulates sum_exp = sum_{neg=1} exp(x - pos), writes log1p(sum_exp).
__global__ void __launch_bounds__(CTL_BS)
ctl_row_kernel(const float* __restrict__ inp,
               const int*   __restrict__ tgt,
               const int*   __restrict__ neg,
               float*       __restrict__ row_loss) {
    const int row = blockIdx.x;
    const int tid = threadIdx.x;

    const int t = tgt[row];                 // block-uniform -> s_load
    if (t < 0) {                            // padding: uniform branch, no WMMA after
        if (tid == 0) row_loss[row] = 0.0f;
        return;
    }

    const size_t base = (size_t)row * CTL_V;
    const float pos = inp[base + (size_t)t];     // uniform gather

    const v4f* ip = (const v4f*)(inp + base);    // 128000 B/row: 16B aligned
    const v4i* np = (const v4i*)(neg + base);

    float part = 0.0f;
    for (int i = tid; i < CTL_V4; i += CTL_BS) {
        v4f x = __builtin_nontemporal_load(ip + i);   // global_load_b128 th:NT
        v4i m = __builtin_nontemporal_load(np + i);
        // Speculative prefetch ~16KB ahead; OOB past buffer end is silently
        // dropped by hardware (TH=0 speculative), so no guard/branch needed.
        __builtin_prefetch((const void*)(ip + i + 4 * CTL_BS), 0, 0);
        __builtin_prefetch((const void*)(np + i + 4 * CTL_BS), 0, 0);
        #pragma unroll
        for (int j = 0; j < 4; ++j) {
            float e = __expf(x[j] - pos);             // v_exp_f32
            part += m[j] ? e : 0.0f;                  // v_cndmask
        }
    }

    // wave32 reduce via WMMA, then combine the 8 waves through LDS.
    const float w = wave_reduce_wmma(part);
    __shared__ float red[CTL_BS / 32];
    const int lane = tid & 31, wid = tid >> 5;
    if (lane == 0) red[wid] = w;
    __syncthreads();
    if (tid == 0) {
        float total = 0.0f;
        #pragma unroll
        for (int i = 0; i < CTL_BS / 32; ++i) total += red[i];
        row_loss[row] = log1pf(total);
    }
}

// Kernel 2: single block folds 2048 row losses + counts non-padding rows.
__global__ void __launch_bounds__(CTL_BS)
ctl_final_kernel(const float* __restrict__ row_loss,
                 const int*   __restrict__ tgt,
                 float*       __restrict__ out) {
    const int tid = threadIdx.x;
    float ls = 0.0f;
    float nc = 0.0f;
    for (int i = tid; i < CTL_ROWS; i += CTL_BS) {
        ls += row_loss[i];
        nc += (tgt[i] != CTL_PAD) ? 1.0f : 0.0f;
    }
    const float lw = wave_reduce_wmma(ls);
    const float nw = wave_reduce_wmma(nc);
    __shared__ float lred[CTL_BS / 32], nred[CTL_BS / 32];
    const int lane = tid & 31, wid = tid >> 5;
    if (lane == 0) { lred[wid] = lw; nred[wid] = nw; }
    __syncthreads();
    if (tid == 0) {
        float L = 0.0f, N = 0.0f;
        #pragma unroll
        for (int i = 0; i < CTL_BS / 32; ++i) { L += lred[i]; N += nred[i]; }
        out[0] = L / N;
    }
}

extern "C" void kernel_launch(void* const* d_in, const int* in_sizes, int n_in,
                              void* d_out, int out_size, void* d_ws, size_t ws_size,
                              hipStream_t stream) {
    const float* inp = (const float*)d_in[0];   // [B,S,V] f32 logits
    const int*   tgt = (const int*)  d_in[1];   // [B,S]   int targets
    const int*   neg = (const int*)  d_in[2];   // [B,S,V] 0/1 mask
    float*       out = (float*)d_out;           // scalar
    float*       row_loss = (float*)d_ws;       // 2048 floats of scratch

    ctl_row_kernel<<<CTL_ROWS, CTL_BS, 0, stream>>>(inp, tgt, neg, row_loss);
    ctl_final_kernel<<<1, CTL_BS, 0, stream>>>(row_loss, tgt, out);
}
